// NicDecoder_16621523436183
// MI455X (gfx1250) — matrix operand order repaired and checked by reading, not compiled, not run on hardware
//
#include <hip/hip_runtime.h>
#include <stdint.h>

typedef __attribute__((ext_vector_type(16))) __bf16 v16bf;
typedef __attribute__((ext_vector_type(8)))  float  v8f;
typedef __attribute__((ext_vector_type(4)))  unsigned int u32x4;

#define VOCAB 32000
#define EMB   512
#define HID   512
#define BATCH 32
#define SEQT  48
#define NGATE 2048      // 4*H
#define MPAD  1536      // 48*32 rows (hs padded to 1536)
#define NWG   8         // workgroups cooperating on the LSTM recurrence
#define HSLICE 64       // hidden units per recurrence workgroup (512/8)

union Frag {
    v16bf v;
    u32x4 q[2];
};

__device__ __forceinline__ uint16_t f2bf(float f) {
    uint32_t u = __float_as_uint(f);
    return (uint16_t)((u + 0x7FFFu + ((u >> 16) & 1u)) >> 16);   // RNE
}
__device__ __forceinline__ float bf2f(uint16_t h) {
    return __uint_as_float(((uint32_t)h) << 16);
}
__device__ __forceinline__ float sigmoidf(float x) { return 1.0f / (1.0f + __expf(-x)); }

__device__ __forceinline__ v8f wmma_bf16(v16bf a, v16bf b, v8f c) {
    return __builtin_amdgcn_wmma_f32_16x16x32_bf16(false, a, false, b, (short)0, c, false, false);
}

// A fragment: 16x32 bf16 tile from row-major [*,512].
// lane l: row m = mrow + (l&15); K pairs at kb+(l>>4)*8 .. +7 and +16..+23
__device__ __forceinline__ Frag load_fragA(const uint16_t* S, int mrow, int kb, int lane) {
    const int m  = mrow + (lane & 15);
    const int ko = kb + ((lane >> 4) << 3);
    const uint16_t* p = S + (size_t)m * 512 + ko;
    Frag f;
    f.q[0] = *(const u32x4*)p;          // K ko..ko+7
    f.q[1] = *(const u32x4*)(p + 16);   // K ko+16..ko+23
    return f;
}

// B fragment: 32x16 tile where B[k][n] = W[n][k], W row-major [N,512].
// lane l: col n = ncol + (l&15); K pairs at kb+(l>>4)*8 .. +7 and +16..+23
__device__ __forceinline__ Frag load_fragB(const uint16_t* W, int ncol, int kb, int lane) {
    const int n = ncol + (lane & 15);
    const int k = kb + ((lane >> 4) << 3);
    const uint16_t* p = W + (size_t)n * 512 + k;
    Frag f;
    f.q[0] = *(const u32x4*)p;
    f.q[1] = *(const u32x4*)(p + 16);
    return f;
}

// ---------------------------------------------------------------------------
// Generic split-bf16 GEMM: out[r+off, col] = A[r,:]·B[col,:] + bias[col]
// A: [MPAD,512] hi/lo bf16 row-major.  B: [ncols,512] hi/lo bf16 row-major.
// Block = 64 M x 256 N, 8 waves (2x4), each wave 32x64 (2x4 WMMA tiles).
// K-loop fully unrolled: no back-edge -> no modulo-scheduled accumulator
// rotation (which costs WMMA->VALU hazard NOPs + v_mov per rotated tile).
// ---------------------------------------------------------------------------
__global__ __launch_bounds__(256) void gemm_bf16split_kernel(
    const uint16_t* __restrict__ Ahi, const uint16_t* __restrict__ Alo,
    const uint16_t* __restrict__ Bhi, const uint16_t* __restrict__ Blo,
    const float* __restrict__ bias, float* __restrict__ out,
    int ncols, int out_row_offset, int mvalid)
{
    const int lane  = threadIdx.x & 31;
    const int wave  = threadIdx.x >> 5;
    const int mbase = blockIdx.y * 64 + (wave >> 2) * 32;
    const int nbase = blockIdx.x * 256 + (wave & 3) * 64;

    v8f acc[2][4];
#pragma unroll
    for (int mt = 0; mt < 2; ++mt)
#pragma unroll
        for (int nt = 0; nt < 4; ++nt) acc[mt][nt] = {};

#pragma unroll
    for (int ks = 0; ks < 16; ++ks) {
        const int kb = ks * 32;
        const Frag a0h = load_fragA(Ahi, mbase,      kb, lane);
        const Frag a0l = load_fragA(Alo, mbase,      kb, lane);
        const Frag a1h = load_fragA(Ahi, mbase + 16, kb, lane);
        const Frag a1l = load_fragA(Alo, mbase + 16, kb, lane);
#pragma unroll
        for (int nt = 0; nt < 4; ++nt) {
            const Frag bh = load_fragB(Bhi, nbase + nt * 16, kb, lane);
            const Frag bl = load_fragB(Blo, nbase + nt * 16, kb, lane);
            acc[0][nt] = wmma_bf16(a0l.v, bh.v, acc[0][nt]);
            acc[1][nt] = wmma_bf16(a1l.v, bh.v, acc[1][nt]);
            acc[0][nt] = wmma_bf16(a0h.v, bl.v, acc[0][nt]);
            acc[1][nt] = wmma_bf16(a1h.v, bl.v, acc[1][nt]);
            acc[0][nt] = wmma_bf16(a0h.v, bh.v, acc[0][nt]);
            acc[1][nt] = wmma_bf16(a1h.v, bh.v, acc[1][nt]);
        }
    }

#pragma unroll
    for (int nt = 0; nt < 4; ++nt) {
        const int col = nbase + nt * 16 + (lane & 15);
        const float bv = bias[col];
#pragma unroll
        for (int mt = 0; mt < 2; ++mt) {
#pragma unroll
            for (int vv = 0; vv < 8; ++vv) {
                const int r = mbase + mt * 16 + vv + ((lane >> 4) << 3);
                if (r < mvalid)
                    out[(size_t)(r + out_row_offset) * ncols + col] = acc[mt][nt][vv] + bv;
            }
        }
    }
}

// ---------------------------------------------------------------------------
// Multi-WGP LSTM recurrence. NWG workgroups; workgroup g owns hidden units
// [g*64, g*64+64) for all four gates. Each of the 32 waves computes exactly
// one 16x16 gate tile per step (wave w: mt=w&1, jb=(w>>1)&3, gate=w>>3).
// h is exchanged through double-buffered global bf16 hi/lo buffers; one
// device-scope atomic phase barrier per step. Gates are staged through LDS
// so i/f/g/o of a unit meet in one thread; cell state c stays in VGPRs.
// Accumulator split into two independent WMMA chains (hi*hi vs cross terms);
// K-loop fully unrolled to avoid pipeliner register rotation.
// ---------------------------------------------------------------------------
__global__ __launch_bounds__(1024) void lstm_mwg_kernel(
    const uint16_t* __restrict__ Whh_hi, const uint16_t* __restrict__ Whh_lo,
    const float* __restrict__ Gin,
    uint16_t* __restrict__ hs_hi, uint16_t* __restrict__ hs_lo,
    uint16_t* __restrict__ h0hi, uint16_t* __restrict__ h0lo,
    uint16_t* __restrict__ h1hi, uint16_t* __restrict__ h1lo,
    unsigned int* __restrict__ ctr)
{
    __shared__ __align__(16) float gbuf[BATCH * 4 * HSLICE];   // 32 KB

    const int tid   = threadIdx.x;
    const int lane  = tid & 31;
    const int wave  = tid >> 5;
    const int hbase = blockIdx.x * HSLICE;

    const int mt   = wave & 1;             // batch half (rows mt*16..mt*16+15)
    const int jb   = (wave >> 1) & 3;      // 16-wide hidden sub-block
    const int gi   = wave >> 3;            // gate index i,f,g,o
    const int gcol = gi * 512 + hbase + jb * 16;       // column in [0,2048)
    const int lcolb = gi * HSLICE + jb * 16;           // column in LDS gate buf

    float cst[2] = {0.0f, 0.0f};           // cell state for units tid, tid+1024

    for (int t = 0; t < SEQT; ++t) {
        const uint16_t* hin_hi  = (t & 1) ? h1hi : h0hi;
        const uint16_t* hin_lo  = (t & 1) ? h1lo : h0lo;
        uint16_t*       hout_hi = (t & 1) ? h0hi : h1hi;
        uint16_t*       hout_lo = (t & 1) ? h0lo : h1lo;

        // chain P seeded with the precomputed input projection (includes bias)
        const float* g = Gin + (size_t)t * BATCH * NGATE;
        v8f accP = {};
#pragma unroll
        for (int vv = 0; vv < 8; ++vv) {
            const int b = mt * 16 + vv + ((lane >> 4) << 3);
            accP[vv] = g[(size_t)b * NGATE + gcol + (lane & 15)];
        }
        v8f accQ = {};                     // chain Q: cross (lo) terms

        // acc += h @ W_hh^T  (split-bf16, K = 512)
#pragma unroll
        for (int ks = 0; ks < 16; ++ks) {
            const int kb = ks * 32;
            const Frag ah = load_fragA(hin_hi, mt * 16, kb, lane);
            const Frag al = load_fragA(hin_lo, mt * 16, kb, lane);
            const Frag bh = load_fragB(Whh_hi, gcol, kb, lane);
            const Frag bl = load_fragB(Whh_lo, gcol, kb, lane);
            accQ = wmma_bf16(al.v, bh.v, accQ);
            accP = wmma_bf16(ah.v, bh.v, accP);
            accQ = wmma_bf16(ah.v, bl.v, accQ);
        }

        // stage pre-activation gates to LDS: gbuf[b][gi*64 + hid_local]
#pragma unroll
        for (int vv = 0; vv < 8; ++vv) {
            const int b = mt * 16 + vv + ((lane >> 4) << 3);
            gbuf[b * (4 * HSLICE) + lcolb + (lane & 15)] = accP[vv] + accQ[vv];
        }
        __syncthreads();

        // cell update: 2 units per thread (u = b*64 + hid_local)
#pragma unroll
        for (int uu = 0; uu < 2; ++uu) {
            const int u   = tid + uu * 1024;
            const int b   = u >> 6;
            const int hid = u & (HSLICE - 1);
            const float* gr = &gbuf[b * (4 * HSLICE)];
            const float iv = sigmoidf(gr[0 * HSLICE + hid]);
            const float fv = sigmoidf(gr[1 * HSLICE + hid]);
            const float gv = tanhf(gr[2 * HSLICE + hid]);
            const float ov = sigmoidf(gr[3 * HSLICE + hid]);
            const float cn = fv * cst[uu] + iv * gv;
            cst[uu] = cn;
            const float hv = ov * tanhf(cn);
            const uint16_t hh = f2bf(hv);
            const uint16_t hl = f2bf(hv - bf2f(hh));
            const int col = hbase + hid;
            hout_hi[b * HID + col] = hh;
            hout_lo[b * HID + col] = hl;
            if (t >= 1) {
                const size_t r = (size_t)(t - 1) * BATCH + b;
                hs_hi[r * HID + col] = hh;
                hs_lo[r * HID + col] = hl;
            }
        }

        // ---- inter-workgroup phase barrier (release h slice, acquire peers') ----
        __threadfence();            // make this thread's h stores device-visible
        __syncthreads();            // all slice stores + gbuf reads done
        if (tid == 0) {
            __hip_atomic_fetch_add(ctr, 1u, __ATOMIC_RELEASE, __HIP_MEMORY_SCOPE_AGENT);
            while (__hip_atomic_load(ctr, __ATOMIC_ACQUIRE, __HIP_MEMORY_SCOPE_AGENT)
                   < (unsigned)(NWG * (t + 1))) {
                __builtin_amdgcn_s_sleep(1);
            }
        }
        __syncthreads();
        __threadfence();            // acquire side: don't read stale peer h
    }
}

// ---------------------------------------------------------------------------
// Prep kernels
// ---------------------------------------------------------------------------
__global__ void convert_split_kernel(const float* __restrict__ src,
                                     uint16_t* __restrict__ hi,
                                     uint16_t* __restrict__ lo, int n)
{
    const int i = blockIdx.x * blockDim.x + threadIdx.x;
    if (i < n) {
        const float x = src[i];
        const uint16_t h = f2bf(x);
        hi[i] = h;
        lo[i] = f2bf(x - bf2f(h));
    }
}

__global__ void build_x_kernel(const float* __restrict__ img, const int* __restrict__ gt,
                               const float* __restrict__ emb,
                               uint16_t* __restrict__ Xhi, uint16_t* __restrict__ Xlo)
{
    const int i = blockIdx.x * blockDim.x + threadIdx.x;
    if (i >= MPAD * EMB) return;
    const int e = i & (EMB - 1);
    const int r = i >> 9;          // t*32 + b
    const int b = r & 31;
    const int t = r >> 5;
    float x;
    if (t == 0) {
        x = img[b * EMB + e];                                    // image_features[0,b,:]
    } else {
        const int tok = (t == 1) ? 1 : gt[b * SEQT + (t - 1)];   // BOS then gt[:,1..46]
        x = emb[(size_t)tok * EMB + e];
    }
    const uint16_t h = f2bf(x);
    Xhi[i] = h;
    Xlo[i] = f2bf(x - bf2f(h));
}

__global__ void misc_prep_kernel(const float* __restrict__ b_ih, const float* __restrict__ b_hh,
                                 float* __restrict__ bias,
                                 uint16_t* __restrict__ hs_hi, uint16_t* __restrict__ hs_lo,
                                 uint16_t* __restrict__ h0hi, uint16_t* __restrict__ h0lo,
                                 unsigned int* __restrict__ ctr,
                                 float* __restrict__ out)
{
    const int i = blockIdx.x * blockDim.x + threadIdx.x;
    if (i == 0) *ctr = 0u;                 // reset phase barrier every launch
    if (i < NGATE) bias[i] = b_ih[i] + b_hh[i];
    if (i < BATCH * HID) {                 // zero hs padding rows + initial h0
        hs_hi[1504 * HID + i] = 0;
        hs_lo[1504 * HID + i] = 0;
        h0hi[i] = 0;
        h0lo[i] = 0;
    }
    if (i < BATCH * VOCAB) out[i] = 0.0f;  // outputs[0] = 0
}

// ---------------------------------------------------------------------------
extern "C" void kernel_launch(void* const* d_in, const int* in_sizes, int n_in,
                              void* d_out, int out_size, void* d_ws, size_t ws_size,
                              hipStream_t stream)
{
    const float* img    = (const float*)d_in[0];   // [1,32,512]
    const int*   gt     = (const int*)  d_in[1];   // [32,48]
    const float* emb    = (const float*)d_in[2];   // [32000,512]
    const float* W_ih   = (const float*)d_in[3];   // [2048,512]
    const float* W_hh   = (const float*)d_in[4];   // [2048,512]
    const float* b_ih   = (const float*)d_in[5];   // [2048]
    const float* b_hh   = (const float*)d_in[6];   // [2048]
    const float* W_head = (const float*)d_in[7];   // [32000,512]
    const float* b_head = (const float*)d_in[8];   // [32000]
    float* out = (float*)d_out;                    // [48,32,32000]
    (void)in_sizes; (void)n_in; (void)out_size; (void)ws_size;

    char* ws = (char*)d_ws;
    size_t off = 0;
    auto take = [&](size_t bytes) -> char* {
        char* p = ws + off;
        off = (off + bytes + 255) & ~(size_t)255;
        return p;
    };
    uint16_t* Whead_hi = (uint16_t*)take((size_t)VOCAB * HID * 2);
    uint16_t* Whead_lo = (uint16_t*)take((size_t)VOCAB * HID * 2);
    uint16_t* Wih_hi   = (uint16_t*)take((size_t)NGATE * EMB * 2);
    uint16_t* Wih_lo   = (uint16_t*)take((size_t)NGATE * EMB * 2);
    uint16_t* Whh_hi   = (uint16_t*)take((size_t)NGATE * HID * 2);
    uint16_t* Whh_lo   = (uint16_t*)take((size_t)NGATE * HID * 2);
    uint16_t* Xhi      = (uint16_t*)take((size_t)MPAD * EMB * 2);
    uint16_t* Xlo      = (uint16_t*)take((size_t)MPAD * EMB * 2);
    float*    Gin      = (float*)   take((size_t)MPAD * NGATE * 4);
    uint16_t* hs_hi    = (uint16_t*)take((size_t)MPAD * HID * 2);
    uint16_t* hs_lo    = (uint16_t*)take((size_t)MPAD * HID * 2);
    uint16_t* h0hi     = (uint16_t*)take((size_t)BATCH * HID * 2);
    uint16_t* h0lo     = (uint16_t*)take((size_t)BATCH * HID * 2);
    uint16_t* h1hi     = (uint16_t*)take((size_t)BATCH * HID * 2);
    uint16_t* h1lo     = (uint16_t*)take((size_t)BATCH * HID * 2);
    float*    bias     = (float*)   take((size_t)NGATE * 4);
    unsigned int* ctr  = (unsigned int*)take(256);

    convert_split_kernel<<<(VOCAB * HID + 255) / 256, 256, 0, stream>>>(
        W_head, Whead_hi, Whead_lo, VOCAB * HID);
    convert_split_kernel<<<(NGATE * EMB + 255) / 256, 256, 0, stream>>>(
        W_ih, Wih_hi, Wih_lo, NGATE * EMB);
    convert_split_kernel<<<(NGATE * HID + 255) / 256, 256, 0, stream>>>(
        W_hh, Whh_hi, Whh_lo, NGATE * HID);
    build_x_kernel<<<(MPAD * EMB + 255) / 256, 256, 0, stream>>>(img, gt, emb, Xhi, Xlo);
    misc_prep_kernel<<<(BATCH * VOCAB + 255) / 256, 256, 0, stream>>>(
        b_ih, b_hh, bias, hs_hi, hs_lo, h0hi, h0lo, ctr, out);

    // Gin[1536,2048] = X @ W_ih^T + (b_ih + b_hh)
    gemm_bf16split_kernel<<<dim3(NGATE / 256, MPAD / 64), 256, 0, stream>>>(
        Xhi, Xlo, Wih_hi, Wih_lo, bias, Gin, NGATE, 0, MPAD);

    // 48-step recurrence on 8 WGPs, produces hs (bf16 hi/lo) rows 0..1503
    lstm_mwg_kernel<<<NWG, 1024, 0, stream>>>(
        Whh_hi, Whh_lo, Gin, hs_hi, hs_lo, h0hi, h0lo, h1hi, h1lo, ctr);

    // logits: out[r+32, col] = hs[r,:]·W_head[col,:] + b_head[col], r < 1504
    gemm_bf16split_kernel<<<dim3(VOCAB / 256, MPAD / 64), 256, 0, stream>>>(
        hs_hi, hs_lo, Whead_hi, Whead_lo, b_head, out, VOCAB, 32, 1504);
}
